// HARMamba_4544075399355
// MI455X (gfx1250) — compile-verified, matrix operand order checked by reading, not run on hardware
//
#include <hip/hip_runtime.h>
#include <math.h>

#define D_MODEL   128
#define D_STATE   16
#define D_CONV    4
#define D_INNER   256
#define DT_RANK   8
#define DEPTH     12
#define NUM_CLASSES 12
#define IN_CH     9
#define SEQ_LEN   1024
#define PATCH     16
#define STRIDE    8
#define PADW      4
#define NUM_PATCHES 128
#define BATCH     16
#define LTOK      (NUM_PATCHES + 1)   /* 129 tokens incl. cls */
#define TOK       (BATCH * LTOK)      /* 2064 */
#define XPROJ_N   (DT_RANK + 2 * D_STATE)  /* 40 */
#define XPROJ_NP  48                  /* padded to multiple of 16 */
#define DT_KP     32                  /* dtproj K padded 8 -> 32 */

typedef _Float16 v16h __attribute__((ext_vector_type(16)));
typedef _Float16 v8h  __attribute__((ext_vector_type(8)));
typedef float    v8f  __attribute__((ext_vector_type(8)));

__device__ __forceinline__ float sigmoidf_(float x) { return 1.0f / (1.0f + __expf(-x)); }
__device__ __forceinline__ float siluf_(float x)    { return x * sigmoidf_(x); }
__device__ __forceinline__ float softplusf_(float x){ return (x > 20.f) ? x : log1pf(__expf(x)); }
__device__ __forceinline__ float geluf_(float x)    { return 0.5f * x * (1.0f + erff(x * 0.70710678118654752f)); }

// ---------------------------------------------------------------------------
// Weight convert + pad:  dst[m, r, c] = (r<src_rows && c<src_cols) ? src : 0
// ---------------------------------------------------------------------------
__global__ void k_cvt_pad(const float* __restrict__ src, _Float16* __restrict__ dst,
                          int nmat, int src_rows, int dst_rows, int src_cols, int dst_cols)
{
    size_t idx = (size_t)blockIdx.x * blockDim.x + threadIdx.x;
    size_t total = (size_t)nmat * dst_rows * dst_cols;
    if (idx >= total) return;
    int c = (int)(idx % dst_cols);
    int r = (int)((idx / dst_cols) % dst_rows);
    int m = (int)(idx / ((size_t)dst_cols * dst_rows));
    float v = (r < src_rows && c < src_cols)
            ? src[((size_t)m * src_rows + r) * src_cols + c] : 0.0f;
    dst[idx] = (_Float16)v;
}

// ---------------------------------------------------------------------------
// WMMA GEMM: out[M,N] = act(A[M,K] @ W[N,K]^T + bias) + residual
// f16 inputs, one wave per 16x16 tile, 8 waves per block.
// All dims padded: M%16==0, N%16==0, K%32==0.  Store guarded by Nstore.
// act: 0=none, 1=softplus, 2=gelu
// ---------------------------------------------------------------------------
__global__ void __launch_bounds__(256)
k_gemm(const _Float16* __restrict__ A, int lda,
       const _Float16* __restrict__ W,
       const float* __restrict__ bias,
       const float* __restrict__ residual, int ldr,
       float* __restrict__ out32, _Float16* __restrict__ out16, int ldo,
       int M, int N, int Nstore, int K, int act)
{
    int tileN = N >> 4;
    int tile  = blockIdx.x * (blockDim.x >> 5) + (threadIdx.x >> 5);
    if (tile >= (M >> 4) * tileN) return;
    int tm = tile / tileN;
    int tn = tile % tileN;
    int lane = threadIdx.x & 31;
    int r16  = lane & 15;
    int hi   = lane >> 4;

    const _Float16* Ap = A + (size_t)(tm * 16 + r16) * lda + hi * 8;
    const _Float16* Wp = W + (size_t)(tn * 16 + r16) * K   + hi * 8;

    v8f acc = {};
    for (int k0 = 0; k0 < K; k0 += 32) {
        v8h al = *(const v8h*)(Ap + k0);
        v8h ah = *(const v8h*)(Ap + k0 + 16);
        v8h bl = *(const v8h*)(Wp + k0);
        v8h bh = *(const v8h*)(Wp + k0 + 16);
        v16h a = __builtin_shufflevector(al, ah, 0,1,2,3,4,5,6,7,8,9,10,11,12,13,14,15);
        v16h b = __builtin_shufflevector(bl, bh, 0,1,2,3,4,5,6,7,8,9,10,11,12,13,14,15);
        acc = __builtin_amdgcn_wmma_f32_16x16x32_f16(false, a, false, b,
                                                     (short)0, acc, false, false);
    }

    int cn = tn * 16 + r16;
    if (cn >= Nstore) return;   // whole-lane predicate only in epilogue
    float bv = bias ? bias[cn] : 0.0f;
#pragma unroll
    for (int v = 0; v < 8; ++v) {
        int cm = tm * 16 + v + (hi << 3);
        float val = acc[v] + bv;
        if (act == 1)      val = softplusf_(val);
        else if (act == 2) val = geluf_(val);
        if (residual) val += residual[(size_t)cm * ldr + cn];
        size_t o = (size_t)cm * ldo + cn;
        if (out32) out32[o] = val;
        if (out16) out16[o] = (_Float16)val;
    }
}

// ---------------------------------------------------------------------------
// Patch embedding: strided conv (stride 8, pad 4, k=16, 9ch->128ch)
// + LayerNorm + pos_embed -> h tokens 1..128.  One block per patch.
// ---------------------------------------------------------------------------
__global__ void k_patch_embed(const float* __restrict__ x,
                              const float* __restrict__ w,
                              const float* __restrict__ cb,
                              const float* __restrict__ g,
                              const float* __restrict__ bb,
                              const float* __restrict__ pos,
                              float* __restrict__ h)
{
    int blk = blockIdx.x;
    int b = blk / NUM_PATCHES, p = blk % NUM_PATCHES;
    int c = threadIdx.x;
    float acc = cb[c];
    int base = p * STRIDE - PADW;
    for (int t = 0; t < PATCH; ++t) {
        int s = base + t;
        if (s >= 0 && s < SEQ_LEN) {
            const float* xp = x + ((size_t)b * SEQ_LEN + s) * IN_CH;
            const float* wp = w + ((size_t)c * IN_CH) * PATCH + t;
#pragma unroll
            for (int ic = 0; ic < IN_CH; ++ic) acc += xp[ic] * wp[ic * PATCH];
        }
    }
    __shared__ float red[D_MODEL];
    red[c] = acc; __syncthreads();
    for (int s2 = 64; s2 > 0; s2 >>= 1) { if (c < s2) red[c] += red[c + s2]; __syncthreads(); }
    float mean = red[0] * (1.0f / D_MODEL);
    __syncthreads();
    float dv = acc - mean;
    red[c] = dv * dv; __syncthreads();
    for (int s2 = 64; s2 > 0; s2 >>= 1) { if (c < s2) red[c] += red[c + s2]; __syncthreads(); }
    float var = red[0] * (1.0f / D_MODEL);
    float outv = dv * rsqrtf(var + 1e-5f) * g[c] + bb[c];
    h[((size_t)b * LTOK + 1 + p) * D_MODEL + c] = outv + pos[(size_t)p * D_MODEL + c];
}

__global__ void k_cls(const float* __restrict__ cls, float* __restrict__ h)
{
    int idx = blockIdx.x * blockDim.x + threadIdx.x;
    if (idx >= BATCH * D_MODEL) return;
    int b = idx / D_MODEL, c = idx % D_MODEL;
    h[((size_t)b * LTOK) * D_MODEL + c] = cls[c];
}

// LayerNorm over 128 channels; reads f32 rows (stride row_stride), writes f16.
__global__ void k_layernorm(const float* __restrict__ in,
                            const float* __restrict__ g,
                            const float* __restrict__ bb,
                            _Float16* __restrict__ out, int row_stride)
{
    int r = blockIdx.x, c = threadIdx.x;
    float v = in[(size_t)r * row_stride + c];
    __shared__ float red[D_MODEL];
    red[c] = v; __syncthreads();
    for (int s2 = 64; s2 > 0; s2 >>= 1) { if (c < s2) red[c] += red[c + s2]; __syncthreads(); }
    float mean = red[0] * (1.0f / D_MODEL);
    __syncthreads();
    float dv = v - mean;
    red[c] = dv * dv; __syncthreads();
    for (int s2 = 64; s2 > 0; s2 >>= 1) { if (c < s2) red[c] += red[c + s2]; __syncthreads(); }
    float var = red[0] * (1.0f / D_MODEL);
    out[(size_t)r * D_MODEL + c] = (_Float16)(dv * rsqrtf(var + 1e-5f) * g[c] + bb[c]);
}

// Depthwise conv (k=4) + SiLU on f16 xr; dir selects causal / anti-causal.
__global__ void k_conv_silu(const _Float16* __restrict__ xr,
                            const float* __restrict__ w,
                            const float* __restrict__ b,
                            _Float16* __restrict__ xi, int L, int dir)
{
    int idx = blockIdx.x * blockDim.x + threadIdx.x;
    if (idx >= BATCH * L * D_INNER) return;
    int d = idx % D_INNER;
    int l = (idx / D_INNER) % L;
    int bb = idx / (D_INNER * L);
    float acc = b[d];
#pragma unroll
    for (int j = 0; j < D_CONV; ++j) {
        int pos = dir ? (l + (D_CONV - 1) - j) : (l - (D_CONV - 1) + j);
        if (pos >= 0 && pos < L)
            acc += w[d * D_CONV + j] * (float)xr[((size_t)bb * L + pos) * (2 * D_INNER) + d];
    }
    xi[(size_t)idx] = (_Float16)siluf_(acc);
}

// Selective scan: one block per batch, 256 threads = channels, 16 states in
// registers, B/C staged in LDS.  dir=1 scans backward (flip-scan-flip).
__global__ void k_scan(const float* __restrict__ delta,
                       const _Float16* __restrict__ xi,
                       const _Float16* __restrict__ xdbl,
                       const float* __restrict__ alog,
                       const float* __restrict__ dpar,
                       float* __restrict__ y, int L, int dir)
{
    int b = blockIdx.x;
    int d = threadIdx.x;
    float Arow[D_STATE];
#pragma unroll
    for (int n = 0; n < D_STATE; ++n) Arow[n] = -__expf(alog[(size_t)d * D_STATE + n]);
    float Dp = dpar[d];
    float hs[D_STATE];
#pragma unroll
    for (int n = 0; n < D_STATE; ++n) hs[n] = 0.0f;
    __shared__ float sBC[2 * D_STATE];
    for (int t = 0; t < L; ++t) {
        int l = dir ? (L - 1 - t) : t;
        size_t tok = (size_t)b * L + l;
        __syncthreads();
        if (d < 2 * D_STATE) sBC[d] = (float)xdbl[tok * XPROJ_NP + DT_RANK + d];
        __syncthreads();
        float dl = delta[tok * D_INNER + d];
        float xv = (float)xi[tok * D_INNER + d];
        float dx = dl * xv;
        float acc = 0.0f;
#pragma unroll
        for (int n = 0; n < D_STATE; ++n) {
            float dA = __expf(dl * Arow[n]);
            hs[n] = dA * hs[n] + dx * sBC[n];
            acc += hs[n] * sBC[D_STATE + n];
        }
        y[tok * D_INNER + d] = acc + Dp * xv;
    }
}

// Gating: gated = y * silu(res),  res = xr[:, 256:512]
__global__ void k_gate(const float* __restrict__ y,
                       const _Float16* __restrict__ xr,
                       _Float16* __restrict__ gated)
{
    int idx = blockIdx.x * blockDim.x + threadIdx.x;
    if (idx >= TOK * D_INNER) return;
    int d = idx % D_INNER;
    int tok = idx / D_INNER;
    float res = (float)xr[(size_t)tok * (2 * D_INNER) + D_INNER + d];
    gated[(size_t)idx] = (_Float16)(y[(size_t)idx] * siluf_(res));
}

// ---------------------------------------------------------------------------
extern "C" void kernel_launch(void* const* d_in, const int* in_sizes, int n_in,
                              void* d_out, int out_size, void* d_ws, size_t ws_size,
                              hipStream_t stream)
{
    (void)in_sizes; (void)n_in; (void)out_size; (void)ws_size;
    const float* x            = (const float*)d_in[0];
    const float* conv_patch_w = (const float*)d_in[1];
    const float* conv_patch_b = (const float*)d_in[2];
    const float* pe_norm_g    = (const float*)d_in[3];
    const float* pe_norm_b    = (const float*)d_in[4];
    const float* pos_embed    = (const float*)d_in[5];
    const float* cls_token    = (const float*)d_in[6];
    const float* blk_norm_g   = (const float*)d_in[7];
    const float* blk_norm_b   = (const float*)d_in[8];
    const float* m_inproj_w   = (const float*)d_in[9];
    const float* m_conv_w     = (const float*)d_in[10];
    const float* m_conv_b     = (const float*)d_in[11];
    const float* m_xproj_w    = (const float*)d_in[12];
    const float* m_dtproj_w   = (const float*)d_in[13];
    const float* m_dtproj_b   = (const float*)d_in[14];
    const float* m_alog       = (const float*)d_in[15];
    const float* m_d          = (const float*)d_in[16];
    const float* m_outproj_w  = (const float*)d_in[17];
    const float* fusion_w     = (const float*)d_in[18];
    const float* fusion_b     = (const float*)d_in[19];
    const float* mix_norm_g   = (const float*)d_in[20];
    const float* mix_norm_b   = (const float*)d_in[21];
    const float* mix_fc1_w    = (const float*)d_in[22];
    const float* mix_fc1_b    = (const float*)d_in[23];
    const float* mix_fc2_w    = (const float*)d_in[24];
    const float* mix_fc2_b    = (const float*)d_in[25];
    const float* norm_g       = (const float*)d_in[26];
    const float* norm_b       = (const float*)d_in[27];
    const float* head_w       = (const float*)d_in[28];
    const float* head_b       = (const float*)d_in[29];

    // ---- workspace carve (256B aligned chunks) ----
    char* wsb = (char*)d_ws;
    auto alloc = [&](size_t bytes) -> char* {
        char* p = wsb; wsb += (bytes + 255) & ~(size_t)255; return p;
    };
    float*     h        = (float*)    alloc((size_t)TOK * D_MODEL * 4);
    float*     delta    = (float*)    alloc((size_t)TOK * D_INNER * 4);
    float*     ybuf     = (float*)    alloc((size_t)TOK * D_INNER * 4);
    _Float16*  hn16     = (_Float16*) alloc((size_t)TOK * D_MODEL * 2);
    _Float16*  xr16     = (_Float16*) alloc((size_t)TOK * 2 * D_INNER * 2);
    _Float16*  xi16     = (_Float16*) alloc((size_t)TOK * D_INNER * 2);
    _Float16*  xdbl16   = (_Float16*) alloc((size_t)TOK * XPROJ_NP * 2);
    _Float16*  gated16  = (_Float16*) alloc((size_t)TOK * D_INNER * 2);
    _Float16*  fused16  = (_Float16*) alloc((size_t)TOK * 2 * D_MODEL * 2);
    _Float16*  mlp16    = (_Float16*) alloc((size_t)TOK * 4 * D_MODEL * 2);
    _Float16*  cls16    = (_Float16*) alloc((size_t)BATCH * D_MODEL * 2);
    // f16 weights (padded)
    _Float16*  wInp     = (_Float16*) alloc((size_t)DEPTH*2 * 512 * 128 * 2);
    _Float16*  wXp      = (_Float16*) alloc((size_t)DEPTH*2 * XPROJ_NP * 256 * 2);
    _Float16*  wDt      = (_Float16*) alloc((size_t)DEPTH*2 * 256 * DT_KP * 2);
    _Float16*  wOut     = (_Float16*) alloc((size_t)DEPTH*2 * 128 * 256 * 2);
    _Float16*  wFus     = (_Float16*) alloc((size_t)DEPTH * 128 * 256 * 2);
    _Float16*  wFc1     = (_Float16*) alloc((size_t)(DEPTH/2) * 512 * 128 * 2);
    _Float16*  wFc2     = (_Float16*) alloc((size_t)(DEPTH/2) * 128 * 512 * 2);
    _Float16*  wHead    = (_Float16*) alloc((size_t)16 * 128 * 2);

    auto cvt = [&](const float* src, _Float16* dst, int nmat,
                   int sr, int dr, int sc, int dc) {
        size_t total = (size_t)nmat * dr * dc;
        k_cvt_pad<<<dim3((unsigned)((total + 255) / 256)), dim3(256), 0, stream>>>(
            src, dst, nmat, sr, dr, sc, dc);
    };
    cvt(m_inproj_w,  wInp,  DEPTH*2, 512, 512, 128, 128);
    cvt(m_xproj_w,   wXp,   DEPTH*2, XPROJ_N, XPROJ_NP, 256, 256);
    cvt(m_dtproj_w,  wDt,   DEPTH*2, 256, 256, DT_RANK, DT_KP);
    cvt(m_outproj_w, wOut,  DEPTH*2, 128, 128, 256, 256);
    cvt(fusion_w,    wFus,  DEPTH,   128, 128, 256, 256);
    cvt(mix_fc1_w,   wFc1,  DEPTH/2, 512, 512, 128, 128);
    cvt(mix_fc2_w,   wFc2,  DEPTH/2, 128, 128, 512, 512);
    cvt(head_w,      wHead, 1,       NUM_CLASSES, 16, 128, 128);

    auto gemm = [&](const _Float16* A, int lda, const _Float16* W,
                    const float* bias, const float* resid, int ldr,
                    float* out32, _Float16* out16, int ldo,
                    int M, int N, int Nstore, int K, int act) {
        int tiles = (M >> 4) * (N >> 4);
        k_gemm<<<dim3((unsigned)((tiles + 7) / 8)), dim3(256), 0, stream>>>(
            A, lda, W, bias, resid, ldr, out32, out16, ldo, M, N, Nstore, K, act);
    };

    // --- stem ---
    k_patch_embed<<<dim3(BATCH * NUM_PATCHES), dim3(D_MODEL), 0, stream>>>(
        x, conv_patch_w, conv_patch_b, pe_norm_g, pe_norm_b, pos_embed, h);
    k_cls<<<dim3((BATCH * D_MODEL + 255) / 256), dim3(256), 0, stream>>>(cls_token, h);

    // --- layers ---
    for (int i = 0; i < DEPTH; ++i) {
        k_layernorm<<<dim3(TOK), dim3(D_MODEL), 0, stream>>>(
            h, blk_norm_g + (size_t)i * D_MODEL, blk_norm_b + (size_t)i * D_MODEL,
            hn16, D_MODEL);

        for (int dir = 0; dir < 2; ++dir) {
            size_t wi = (size_t)i * 2 + dir;
            // inproj: [TOK,128] -> xr16 [TOK,512]
            gemm(hn16, D_MODEL, wInp + wi * 512 * 128, nullptr, nullptr, 0,
                 nullptr, xr16, 2 * D_INNER, TOK, 2 * D_INNER, 2 * D_INNER, D_MODEL, 0);
            // depthwise conv + silu (direction-aware)
            k_conv_silu<<<dim3((TOK * D_INNER + 255) / 256), dim3(256), 0, stream>>>(
                xr16, m_conv_w + wi * D_INNER * D_CONV, m_conv_b + wi * D_INNER,
                xi16, LTOK, dir);
            // xproj: [TOK,256] -> xdbl16 [TOK,48]  (dt | B | C | pad)
            gemm(xi16, D_INNER, wXp + wi * XPROJ_NP * 256, nullptr, nullptr, 0,
                 nullptr, xdbl16, XPROJ_NP, TOK, XPROJ_NP, XPROJ_NP, D_INNER, 0);
            // dtproj + softplus: [TOK, 32(pad of 8)] -> delta [TOK,256] (f32)
            gemm(xdbl16, XPROJ_NP, wDt + wi * 256 * DT_KP,
                 m_dtproj_b + wi * D_INNER, nullptr, 0,
                 delta, nullptr, D_INNER, TOK, D_INNER, D_INNER, DT_KP, 1);
            // selective scan
            k_scan<<<dim3(BATCH), dim3(D_INNER), 0, stream>>>(
                delta, xi16, xdbl16, m_alog + wi * D_INNER * D_STATE,
                m_d + wi * D_INNER, ybuf, LTOK, dir);
            // gate with silu(res)
            k_gate<<<dim3((TOK * D_INNER + 255) / 256), dim3(256), 0, stream>>>(
                ybuf, xr16, gated16);
            // outproj -> columns [dir*128, dir*128+128) of fused16
            gemm(gated16, D_INNER, wOut + wi * 128 * 256, nullptr, nullptr, 0,
                 nullptr, fused16 + dir * D_MODEL, 2 * D_MODEL,
                 TOK, D_MODEL, D_MODEL, D_INNER, 0);
        }
        // fusion: h += fused @ W^T + b
        gemm(fused16, 2 * D_MODEL, wFus + (size_t)i * 128 * 256,
             fusion_b + (size_t)i * D_MODEL, h, D_MODEL,
             h, nullptr, D_MODEL, TOK, D_MODEL, D_MODEL, 2 * D_MODEL, 0);

        if (i & 1) {
            size_t j = (size_t)(i / 2);
            k_layernorm<<<dim3(TOK), dim3(D_MODEL), 0, stream>>>(
                h, mix_norm_g + j * D_MODEL, mix_norm_b + j * D_MODEL, hn16, D_MODEL);
            // fc1 + gelu: 128 -> 512 (f16 out)
            gemm(hn16, D_MODEL, wFc1 + j * 512 * 128,
                 mix_fc1_b + j * 4 * D_MODEL, nullptr, 0,
                 nullptr, mlp16, 4 * D_MODEL, TOK, 4 * D_MODEL, 4 * D_MODEL, D_MODEL, 2);
            // fc2 + residual: 512 -> 128 (f32 into h)
            gemm(mlp16, 4 * D_MODEL, wFc2 + j * 128 * 512,
                 mix_fc2_b + j * D_MODEL, h, D_MODEL,
                 h, nullptr, D_MODEL, TOK, D_MODEL, D_MODEL, 4 * D_MODEL, 0);
        }
    }

    // --- final norm (cls rows of h) + head ---
    k_layernorm<<<dim3(BATCH), dim3(D_MODEL), 0, stream>>>(
        h, norm_g, norm_b, cls16, LTOK * D_MODEL);
    gemm(cls16, D_MODEL, wHead, head_b, nullptr, 0,
         (float*)d_out, nullptr, NUM_CLASSES, BATCH /*M=16*/, 16, NUM_CLASSES, D_MODEL, 0);
}
